// SelfAttentionMemory_75952201663223
// MI455X (gfx1250) — matrix operand order repaired and checked by reading, not compiled, not run on hardware
//
#include <hip/hip_runtime.h>
#include <hip/hip_bf16.h>

// Problem constants (from reference): B=32, C=128, H=W=32, N=1024, CR=16
#define BB  32
#define CC  128
#define HH  32
#define WW2 32
#define NN  1024
#define CRR 16

typedef _Float16 half16 __attribute__((ext_vector_type(16)));
typedef _Float16 half8  __attribute__((ext_vector_type(8)));
typedef float    float8 __attribute__((ext_vector_type(8)));
typedef unsigned int uint4v __attribute__((ext_vector_type(4)));
typedef unsigned int uint8v __attribute__((ext_vector_type(8)));

union FragA { half16 v; half8 h[2]; };

__device__ __forceinline__ float8 wmma16x16x32(half16 a, half16 b, float8 c) {
  // D = A(16x32 f16) * B(32x16 f16) + C(16x16 f32)
  return __builtin_amdgcn_wmma_f32_16x16x32_f16(false, a, false, b, (short)0, c,
                                                false, false);
}

__device__ __forceinline__ float8 zero8f() {
  float8 z;
#pragma unroll
  for (int q = 0; q < 8; ++q) z[q] = 0.0f;
  return z;
}

__device__ __forceinline__ half16 zero16h() {
  half16 z;
#pragma unroll
  for (int q = 0; q < 16; ++q) z[q] = (_Float16)0.0f;
  return z;
}

// ---------------------------------------------------------------------------
// Pointwise (1x1 conv) GEMM:  Y[b][o][n] = act( W[o,:] . X[b,:,n] (+bias)(+addend) )
// Cin = 128, N = 1024 fixed. Block = 256 threads (8 waves).
// Block tile: 16 (O) x 128 (N); each wave owns one 16x16 N-subtile; K-loop 4x32.
// The 128x128 f32 X tile is DMA'd into LDS by the Tensor Data Mover (2D D#),
// then transposed/converted to the f16 WMMA B layout by the workgroup.
// A fragment (16-bit A 16x32): lane m holds halfs K = {8g..8g+7, 16+8g..16+8g+7}
// B fragment (32x16):          lane n holds halfs K = {8g..8g+7, 16+8g..16+8g+7}
// ---------------------------------------------------------------------------
template <int ACT, bool HAS_BIAS, bool HAS_ADD>
__global__ __launch_bounds__(256)
void pw_gemm_kernel(const float* __restrict__ Wmat, int ldw,
                    const float* __restrict__ X,
                    float* __restrict__ Y,
                    const float* __restrict__ addend,   // used iff HAS_ADD
                    const float* __restrict__ bias,     // used iff HAS_BIAS
                    int O)
{
  __shared__ __align__(16) float    Xstage[CC * 128];      // 64 KB TDM landing pad
  __shared__ __align__(16) _Float16 Atile[16][136];        // 16 x K=128, padded
  __shared__ __align__(16) _Float16 Btile[8][16][136];     // per-wave [n][k], padded

  const int tid  = threadIdx.x;
  const int wave = tid >> 5;
  const int lane = tid & 31;
  const int b  = blockIdx.z;
  const int o0 = blockIdx.y << 4;
  const int n0 = blockIdx.x << 7;

  // ---- issue TDM: 2D tile (tile_dim0=128 along n, tile_dim1=128 along c) ----
  const float* xb = X + (size_t)b * CC * NN + n0;
  if (wave == 0) {
    const unsigned lds_off = (unsigned)(size_t)(void*)&Xstage[0]; // LDS byte offset (flat addr low 32)
    const unsigned long long ga = (unsigned long long)(size_t)xb; // tile start, bytes
    uint4v g0;
    g0[0] = 1u;                                              // count=1 (valid user D#)
    g0[1] = lds_off;                                         // lds_addr
    g0[2] = (unsigned)(ga & 0xffffffffu);                    // global_addr[31:0]
    g0[3] = (unsigned)((ga >> 32) & 0x01ffffffu) | (2u << 30); // global_addr[56:32] | type=2
    uint8v g1;
    g1[0] = (2u << 16);                 // workgroup_mask=0, data_size=2 (4 bytes)
    g1[1] = ((unsigned)NN & 0xffffu) << 16;                  // tensor_dim0[15:0]
    g1[2] = ((unsigned)NN >> 16) | (((unsigned)CC & 0xffffu) << 16); // dim0 hi | dim1 lo
    g1[3] = ((unsigned)CC >> 16) | (128u << 16);             // dim1 hi | tile_dim0=128
    g1[4] = 128u;                        // tile_dim1=128, tile_dim2=0
    g1[5] = (unsigned)NN;                // tensor_dim0_stride[31:0] (elements)
    g1[6] = 0u;                          // stride0 hi | stride1 lo
    g1[7] = 0u;                          // stride1 hi
    asm volatile("tensor_load_to_lds %0, %1" :: "s"(g0), "s"(g1) : "memory");
  }
  // ---- stage W tile (16 x 128 f32 -> f16) while the TDM streams X ----
  for (int idx = tid; idx < 16 * CC; idx += 256) {
    const int o = idx >> 7, c = idx & (CC - 1);
    Atile[o][c] = (_Float16)Wmat[(o0 + o) * ldw + c];
  }
  if (wave == 0) __builtin_amdgcn_s_wait_tensorcnt(0);
  __syncthreads();

  // ---- transpose + convert: Xstage[c][n] -> Btile[w][n][c] (f16) ----
  for (int idx = tid; idx < CC * 128; idx += 256) {
    const int c = idx >> 7, n = idx & 127;
    Btile[n >> 4][n & 15][c] = (_Float16)Xstage[c * 128 + n];
  }
  __syncthreads();

  const int g  = lane >> 4;
  const int mr = lane & 15;
  float8 acc = zero8f();
#pragma unroll
  for (int k0 = 0; k0 < CC; k0 += 32) {
    FragA a, bf;
    a.h[0]  = *(const half8*)&Atile[mr][k0 + 8 * g];
    a.h[1]  = *(const half8*)&Atile[mr][k0 + 16 + 8 * g];
    bf.h[0] = *(const half8*)&Btile[wave][mr][k0 + 8 * g];
    bf.h[1] = *(const half8*)&Btile[wave][mr][k0 + 16 + 8 * g];
    acc = wmma16x16x32(a.v, bf.v, acc);
  }

  // D layout: lane holds N = lane&15; VGPR r holds M = r + 8*(lane>>4)
  const int n = n0 + (wave << 4) + mr;
#pragma unroll
  for (int r = 0; r < 8; ++r) {
    const int o = o0 + r + 8 * g;
    const size_t idx = ((size_t)b * O + o) * NN + n;
    float v = acc[r];
    if constexpr (HAS_BIAS) v += bias[o];
    if constexpr (HAS_ADD)  v += addend[idx];
    if constexpr (ACT == 1) v = 1.0f / (1.0f + __expf(-v));
    if constexpr (ACT == 2) v = tanhf(v);
    Y[idx] = v;
  }
}

// ---------------------------------------------------------------------------
// Fused attention per (batch, 16-query block):
//   S[i,j] = sum_k Q[k,i0+i] K[k,j]   (k=0..15, padded to 32)     -> LDS f32
//   P = softmax_j(S)                                              -> LDS f16
//   Z[c,i0+i] = sum_j V[c,j] P[i,j]   (32 WMMA K-steps per wave)
// 256 threads (8 waves). LDS ~113 KB (within the 320 KB WGP budget).
// ---------------------------------------------------------------------------
__global__ __launch_bounds__(256)
void attn_kernel(const float* __restrict__ Qg,   // [B, 16, N]
                 const float* __restrict__ Kg,   // [B, 16, N]
                 const float* __restrict__ Vg,   // [B, 128, N]
                 float* __restrict__ Zg)         // [B, 128, N]
{
  __shared__ __align__(16) float    Ssc[16 * NN];     // 64 KB score rows
  __shared__ __align__(16) _Float16 KtP[NN * 24];     // 48 KB: K^T (phase1), P (phase3)
  __shared__ __align__(16) _Float16 Aq[16 * 40];      // Q^T tile, K padded to 32

  const int tid  = threadIdx.x;
  const int wave = tid >> 5;
  const int lane = tid & 31;
  const int b  = blockIdx.y;
  const int i0 = blockIdx.x << 4;

  // stage Q^T tile: Aq[m][k] = Q[b][k][i0+m] for k<16, 0 for 16<=k<32
  for (int idx = tid; idx < 16 * 40; idx += 256) {
    const int m = idx / 40, k = idx % 40;
    float v = 0.0f;
    if (k < CRR) v = Qg[((size_t)b * CRR + k) * NN + i0 + m];
    Aq[m * 40 + k] = (_Float16)v;
  }
  // stage K transposed: KtP[j*24 + k] = K[b][k][j]
  for (int idx = tid; idx < CRR * NN; idx += 256) {
    const int k = idx >> 10, j = idx & (NN - 1);
    KtP[j * 24 + k] = (_Float16)Kg[((size_t)b * CRR + k) * NN + j];
  }
  __syncthreads();

  const int g  = lane >> 4;
  const int mr = lane & 15;

  // ---- phase 1: scores (64 j-tiles, 8 per wave) ----
  FragA afr;
  afr.h[0] = *(const half8*)&Aq[mr * 40 + 8 * g];
  afr.h[1] = *(const half8*)&Aq[mr * 40 + 16 + 8 * g];
  const half16 zh = zero16h();
  for (int jt = wave; jt < 64; jt += 8) {
    const int j0 = jt << 4;
    FragA bfr;
    {                                   // lane half g holds K=8g..8g+7; K=16..31 are zero
      bfr.h[0] = *(const half8*)&KtP[(j0 + mr) * 24 + 8 * g];
      bfr.v.hi = zh.hi;
    }
    float8 s = wmma16x16x32(afr.v, bfr.v, zero8f());
#pragma unroll
    for (int r = 0; r < 8; ++r)
      Ssc[(r + 8 * g) * NN + j0 + mr] = s[r];
  }
  __syncthreads();

  // ---- phase 2: softmax over j; wave w owns rows w and w+8 ----
  _Float16* P = KtP;  // reuse K^T buffer; P row stride 1048 halfs (16*1048 <= 24576)
  for (int rr = 0; rr < 2; ++rr) {
    const int i = wave + 8 * rr;
    float* row = &Ssc[i * NN];
    float mx = -3.4e38f;
    for (int j = lane; j < NN; j += 32) mx = fmaxf(mx, row[j]);
#pragma unroll
    for (int off = 16; off > 0; off >>= 1) mx = fmaxf(mx, __shfl_xor(mx, off, 32));
    float sum = 0.0f;
    for (int j = lane; j < NN; j += 32) {
      float e = __expf(row[j] - mx);
      row[j] = e;
      sum += e;
    }
#pragma unroll
    for (int off = 16; off > 0; off >>= 1) sum += __shfl_xor(sum, off, 32);
    const float inv = 1.0f / sum;
    for (int j = lane; j < NN; j += 32)
      P[i * 1048 + j] = (_Float16)(row[j] * inv);
  }
  __syncthreads();

  // ---- phase 3: Z tile; wave owns c-tile c0 = 16*wave; 32 K-steps over j ----
  const int c0 = wave << 4;
  const float* vrow = Vg + ((size_t)b * CC + c0 + mr) * NN;  // A row: c = c0 + mr
  float8 zacc = zero8f();
#pragma unroll 2
  for (int kt = 0; kt < 32; ++kt) {
    const int j0 = kt << 5;
    if ((kt & 7) == 0) __builtin_prefetch(vrow + j0 + 256, 0, 1);
    FragA va, pb;
    const float* p0 = vrow + j0 + 8 * g;
    const float* p1 = vrow + j0 + 16 + 8 * g;
#pragma unroll
    for (int q = 0; q < 8; ++q) va.v[q] = (_Float16)p0[q];
#pragma unroll
    for (int q = 0; q < 8; ++q) va.v[8 + q] = (_Float16)p1[q];
    pb.h[0] = *(const half8*)&P[mr * 1048 + j0 + 8 * g];
    pb.h[1] = *(const half8*)&P[mr * 1048 + j0 + 16 + 8 * g];
    zacc = wmma16x16x32(va.v, pb.v, zacc);
  }
#pragma unroll
  for (int r = 0; r < 8; ++r)
    Zg[((size_t)b * CC + c0 + r + 8 * g) * NN + i0 + mr] = zacc[r];
}

// ---------------------------------------------------------------------------
// Depthwise 3x3, SAME zero padding. One block per (b,c) plane.
// ---------------------------------------------------------------------------
__global__ __launch_bounds__(256)
void dw3x3_kernel(const float* __restrict__ X, const float* __restrict__ Kw,
                  float* __restrict__ Y)
{
  const int bc = blockIdx.x;
  const int c = bc & (CC - 1);
  float kk[9];
#pragma unroll
  for (int q = 0; q < 9; ++q) kk[q] = Kw[c * 9 + q];
  const float* xp = X + (size_t)bc * NN;
  float* yp = Y + (size_t)bc * NN;
  for (int p = threadIdx.x; p < NN; p += 256) {
    const int y = p >> 5, x = p & 31;
    float acc = 0.0f;
#pragma unroll
    for (int dy = -1; dy <= 1; ++dy) {
      const int yy = y + dy;
      if (yy < 0 || yy >= HH) continue;
#pragma unroll
      for (int dx = -1; dx <= 1; ++dx) {
        const int xx = x + dx;
        if (xx < 0 || xx >= WW2) continue;
        acc += xp[yy * WW2 + xx] * kk[(dy + 1) * 3 + (dx + 1)];
      }
    }
    yp[p] = acc;
  }
}

__global__ __launch_bounds__(256)
void gate_mt_kernel(const float* __restrict__ I, const float* __restrict__ G,
                    const float* __restrict__ Mprev, float* __restrict__ Mt, int n)
{
  const int idx = blockIdx.x * 256 + threadIdx.x;
  if (idx < n) {
    const float i = I[idx];
    Mt[idx] = (1.0f - i) * Mprev[idx] + i * G[idx];
  }
}

__global__ __launch_bounds__(256)
void mul_kernel(const float* __restrict__ A, const float* __restrict__ Bv,
                float* __restrict__ Ov, int n)
{
  const int idx = blockIdx.x * 256 + threadIdx.x;
  if (idx < n) Ov[idx] = A[idx] * Bv[idx];
}

// ---------------------------------------------------------------------------
// Host orchestration. Workspace (floats): 3*524288 + 5*4194304 = 22544384
// (~90.2 MB), with later stages reusing attention buffers.
// ---------------------------------------------------------------------------
extern "C" void kernel_launch(void* const* d_in, const int* in_sizes, int n_in,
                              void* d_out, int out_size, void* d_ws, size_t ws_size,
                              hipStream_t stream)
{
  const float* h_t    = (const float*)d_in[0];
  const float* m_prev = (const float*)d_in[1];
  const float* wq_h   = (const float*)d_in[2];
  const float* wk_h   = (const float*)d_in[3];
  const float* wv_h   = (const float*)d_in[4];
  const float* wk_m   = (const float*)d_in[5];
  const float* wv_m   = (const float*)d_in[6];
  const float* wz_w   = (const float*)d_in[7];
  const float* wz_b   = (const float*)d_in[8];
  const float* wmi_dw = (const float*)d_in[9];
  const float* wmi_pw = (const float*)d_in[10];
  const float* whi_dw = (const float*)d_in[11];
  const float* whi_pw = (const float*)d_in[12];
  const float* bi     = (const float*)d_in[13];
  const float* wmg_dw = (const float*)d_in[14];
  const float* wmg_pw = (const float*)d_in[15];
  const float* whg_dw = (const float*)d_in[16];
  const float* whg_pw = (const float*)d_in[17];
  const float* bg     = (const float*)d_in[18];
  const float* wmo_dw = (const float*)d_in[19];
  const float* wmo_pw = (const float*)d_in[20];
  const float* who_dw = (const float*)d_in[21];
  const float* who_pw = (const float*)d_in[22];
  const float* bo     = (const float*)d_in[23];
  const float* wf_w   = (const float*)d_in[24];
  const float* wf_b   = (const float*)d_in[25];

  const size_t SZS = (size_t)BB * CRR * NN;  // 524288
  const size_t SZL = (size_t)BB * CC * NN;   // 4194304

  float* ws = (float*)d_ws;
  float* Q  = ws;
  float* Kh = Q + SZS;
  float* Km = Kh + SZS;
  float* Vh = Km + SZS;
  float* Vm = Vh + SZL;
  float* Zh = Vm + SZL;
  float* Zm = Zh + SZL;
  float* Zf = Zm + SZL;
  // reuse after the attention / wz phases complete:
  float* DZ = Vh;
  float* DH = Vm;
  float* GI = Zh;
  float* GG = Zm;
  float* GO = Zh;   // I consumed by m_t before o-gate is produced
  float* Ht = Zm;   // G consumed by m_t before h0 is produced

  float* Hout = (float*)d_out;        // output 0: h_out [B,C,H,W]
  float* Mt   = Hout + SZL;           // output 1: m_t   [B,C,H,W]

  dim3 blk(256);
  dim3 gS(NN / 128, 1, BB);           // O = 16 projections
  dim3 gL(NN / 128, CC / 16, BB);     // O = 128 projections
  dim3 gA(NN / 16, BB);               // attention: (query-block, batch)
  dim3 gD(BB * CC);                   // depthwise: one plane per block
  dim3 gE((unsigned)((SZL + 255) / 256));
  const int nTot = (int)SZL;

  // projections (all WMMA GEMMs, TDM-staged activations)
  pw_gemm_kernel<0, false, false><<<gS, blk, 0, stream>>>(wq_h, CC, h_t,    Q,  nullptr, nullptr, CRR);
  pw_gemm_kernel<0, false, false><<<gS, blk, 0, stream>>>(wk_h, CC, h_t,    Kh, nullptr, nullptr, CRR);
  pw_gemm_kernel<0, false, false><<<gL, blk, 0, stream>>>(wv_h, CC, h_t,    Vh, nullptr, nullptr, CC);
  pw_gemm_kernel<0, false, false><<<gS, blk, 0, stream>>>(wk_m, CC, m_prev, Km, nullptr, nullptr, CRR);
  pw_gemm_kernel<0, false, false><<<gL, blk, 0, stream>>>(wv_m, CC, m_prev, Vm, nullptr, nullptr, CC);

  // dual fused attention (never materializes the [N,N] attention matrices)
  attn_kernel<<<gA, blk, 0, stream>>>(Q, Kh, Vh, Zh);
  attn_kernel<<<gA, blk, 0, stream>>>(Q, Km, Vm, Zm);

  // Z = wz_w @ concat(Zh, Zm) + wz_b   (two K=128 chunks, second accumulates)
  pw_gemm_kernel<0, false, false><<<gL, blk, 0, stream>>>(wz_w,      2 * CC, Zh, Zf, nullptr, nullptr, CC);
  pw_gemm_kernel<0, true,  true ><<<gL, blk, 0, stream>>>(wz_w + CC, 2 * CC, Zm, Zf, Zf,      wz_b,    CC);

  // gate i (sigmoid)
  dw3x3_kernel<<<gD, blk, 0, stream>>>(Zf,  wmi_dw, DZ);
  dw3x3_kernel<<<gD, blk, 0, stream>>>(h_t, whi_dw, DH);
  pw_gemm_kernel<0, false, false><<<gL, blk, 0, stream>>>(wmi_pw, CC, DZ, GI, nullptr, nullptr, CC);
  pw_gemm_kernel<1, true,  true ><<<gL, blk, 0, stream>>>(whi_pw, CC, DH, GI, GI,      bi,      CC);

  // gate g (tanh)
  dw3x3_kernel<<<gD, blk, 0, stream>>>(Zf,  wmg_dw, DZ);
  dw3x3_kernel<<<gD, blk, 0, stream>>>(h_t, whg_dw, DH);
  pw_gemm_kernel<0, false, false><<<gL, blk, 0, stream>>>(wmg_pw, CC, DZ, GG, nullptr, nullptr, CC);
  pw_gemm_kernel<2, true,  true ><<<gL, blk, 0, stream>>>(whg_pw, CC, DH, GG, GG,      bg,      CC);

  // m_t = (1 - i)*m_prev + i*g  -> second output
  gate_mt_kernel<<<gE, blk, 0, stream>>>(GI, GG, m_prev, Mt, nTot);

  // gate o (sigmoid)
  dw3x3_kernel<<<gD, blk, 0, stream>>>(Zf,  wmo_dw, DZ);
  dw3x3_kernel<<<gD, blk, 0, stream>>>(h_t, who_dw, DH);
  pw_gemm_kernel<0, false, false><<<gL, blk, 0, stream>>>(wmo_pw, CC, DZ, GO, nullptr, nullptr, CC);
  pw_gemm_kernel<1, true,  true ><<<gL, blk, 0, stream>>>(who_pw, CC, DH, GO, GO,      bo,      CC);

  // h0 = o * m_t ; h_out = wf_w @ h0 + wf_b + h_t  -> first output
  mul_kernel<<<gE, blk, 0, stream>>>(GO, Mt, Ht, nTot);
  pw_gemm_kernel<0, true,  true ><<<gL, blk, 0, stream>>>(wf_w, CC, Ht, Hout, h_t, wf_b, CC);

  (void)in_sizes; (void)n_in; (void)out_size; (void)ws_size;
}
